// BaseVoxelBackbone_37984690766000
// MI455X (gfx1250) — matrix operand order repaired and checked by reading, not compiled, not run on hardware
//
#include <hip/hip_runtime.h>
#include <math.h>

typedef __attribute__((ext_vector_type(16))) __bf16 v16bf;
typedef __attribute__((ext_vector_type(8)))  float  v8f;
typedef __attribute__((ext_vector_type(4)))  unsigned int v4u;

union FragB16 { v4u q[2]; v16bf v; };

// voxel grid geometry (from reference)
#define NV0 (2*41*200*176)   // 2,886,400
#define NV1 (2*21*100*88)    //   369,600
#define NV2 (2*11*50*44)     //    48,400
#define NV3 (2*5*25*22)      //     5,500
#define NV4 (2*2*25*22)      //     2,200

// ---------------- voxelization (deterministic fixed-point atomics) ----------------
__global__ void voxelize_kernel(const float* __restrict__ pts,
                                const float* __restrict__ feats,
                                int* __restrict__ sums,
                                unsigned int* __restrict__ cnt, int npts)
{
  int t = blockIdx.x * blockDim.x + threadIdx.x;
  int pt = t >> 4, c = t & 15;
  if (pt >= npts) return;
  float bf = pts[(size_t)pt*4 + 0];
  float px = pts[(size_t)pt*4 + 1];
  float py = pts[(size_t)pt*4 + 2];
  float pz = pts[(size_t)pt*4 + 3];
  int b  = (int)bf;
  int vx = (int)floorf((px - 0.0f)  / 0.4f);
  int vy = (int)floorf((py + 40.0f) / 0.4f);
  int vz = (int)floorf((pz + 3.0f)  / 0.1f);
  if (vx < 0 || vy < 0 || vz < 0 || vx >= 176 || vy >= 200 || vz >= 40 || b < 0 || b >= 2) return;
  size_t flat = ((size_t)((b*41 + vz)*200 + vy))*176 + vx;
  int q = __float2int_rn(feats[(size_t)pt*16 + c] * 65536.0f);
  atomicAdd(&sums[flat*16 + c], q);
  if (c == 0) atomicAdd(&cnt[flat], 1u);
}

__global__ void finalize_kernel(const int* __restrict__ sums,
                                const unsigned int* __restrict__ cnt,
                                __bf16* __restrict__ act,
                                unsigned char* __restrict__ mask, int nvox)
{
  int t = blockIdx.x * blockDim.x + threadIdx.x;
  int v = t >> 4, c = t & 15;
  if (v >= nvox) return;
  unsigned int cn = cnt[v];
  float denom = (float)(cn > 0u ? cn : 1u);
  float val = ((float)sums[(size_t)v*16 + c] * (1.0f/65536.0f)) / denom;
  act[(size_t)v*16 + c] = (__bf16)val;
  if (c == 0) mask[v] = cn > 0u ? 1 : 0;
}

// ---------------- weight repack: f32 (kz,ky,kx,cin,cout) -> bf16 [kz*KH+ky][cout][Kpad] ----------------
template<int CIN,int COUT,int KD,int KH,int KW>
__global__ void prep_w(const float* __restrict__ w, __bf16* __restrict__ wb)
{
  constexpr int KPAD = ((KW*CIN + 31)/32)*32;
  const int total = KD*KH*COUT*KPAD;
  int t = blockIdx.x * blockDim.x + threadIdx.x;
  if (t >= total) return;
  int k    = t % KPAD;
  int co   = (t / KPAD) % COUT;
  int kzky = t / (KPAD*COUT);
  int kz = kzky / KH, ky = kzky % KH;
  float v = 0.0f;
  if (k < KW*CIN) {
    int kx = k / CIN, c = k % CIN;
    v = w[(size_t)(((kz*KH + ky)*KW + kx)*CIN + c) * COUT + co];
  }
  wb[t] = (__bf16)v;
}

// ---------------- sparse-layer mask propagation: OR over receptive field ----------------
template<int KD,int KH,int KW,int SZ,int SY,int SX,int PZ,int PY,int PX,
         int DIN,int HIN,int WIN,int DOUT,int HOUT,int WOUT>
__global__ void mask_conv(const unsigned char* __restrict__ mi, unsigned char* __restrict__ mo)
{
  const int total = 2*DOUT*HOUT*WOUT;
  int t = blockIdx.x * blockDim.x + threadIdx.x;
  if (t >= total) return;
  int x = t % WOUT;
  int y = (t / WOUT) % HOUT;
  int z = (t / (WOUT*HOUT)) % DOUT;
  int b = t / (WOUT*HOUT*DOUT);
  unsigned char m = 0;
  for (int kz = 0; kz < KD; ++kz) {
    int zi = z*SZ + kz - PZ; if (zi < 0 || zi >= DIN) continue;
    for (int ky = 0; ky < KH; ++ky) {
      int yi = y*SY + ky - PY; if (yi < 0 || yi >= HIN) continue;
      for (int kx = 0; kx < KW; ++kx) {
        int xi = x*SX + kx - PX; if (xi < 0 || xi >= WIN) continue;
        m |= mi[(size_t)((b*DIN + zi)*HIN + yi)*WIN + xi];
      }
    }
  }
  mo[t] = m ? 1 : 0;
}

// ---------------- implicit-GEMM conv via v_wmma_f32_16x16x32_bf16 ----------------
// One wave computes MT m-tiles (MT*16 output voxels along W) x all NT cout-tiles.
// A-fragments (from LDS row slab) are reused across NT; B-fragments (packed bf16
// weights) are reused across MT -> MT*NT WMMAs per 32-wide K chunk.
template<int CIN,int COUT,int KD,int KH,int KW,
         int SZ,int SY,int SX,int PZ,int PY,int PX,
         int DIN,int HIN,int WIN,int DOUT,int HOUT,int WOUT,int MT>
__global__ __launch_bounds__(32)
void conv_wmma(const __bf16* __restrict__ act_in,
               __bf16* __restrict__ act_out,
               const __bf16* __restrict__ wb,
               const float* __restrict__ scale,
               const float* __restrict__ bias,
               const unsigned char* __restrict__ omask)
{
  constexpr int KPAD = ((KW*CIN + 31)/32)*32;       // K per (kz,ky), zero-padded to x32
  constexpr int NT   = COUT/16;
  constexpr int NXP  = (MT*16 - 1)*SX + KPAD/CIN;   // x-positions so A-reads stay in-bounds
  __shared__ __bf16 row[NXP*CIN];

  const int lane = threadIdx.x;
  const int hi = lane >> 4, lo = lane & 15;
  const int xtw = blockIdx.x;                        // wave-tile along W (covers MT*16 voxels)
  const int zo = blockIdx.y / HOUT, yo = blockIdx.y % HOUT;
  const int b  = blockIdx.z;
  const int x0 = xtw*MT*16*SX - PX;

  v8f acc[MT][NT];
#pragma unroll
  for (int mt = 0; mt < MT; ++mt)
#pragma unroll
    for (int nt = 0; nt < NT; ++nt)
      acc[mt][nt] = (v8f){};

  for (int kz = 0; kz < KD; ++kz) {
    const int zi = zo*SZ + kz - PZ;
    if (zi < 0 || zi >= DIN) continue;
    for (int ky = 0; ky < KH; ++ky) {
      const int yi = yo*SY + ky - PY;
      if (yi < 0 || yi >= HIN) continue;

      // cooperatively stage the input row slab [x0 .. x0+NXP) x CIN into LDS (zero OOB)
      const unsigned int* src =
        (const unsigned int*)(act_in + (size_t)((b*DIN + zi)*HIN + yi) * WIN * CIN);
      unsigned int* dstu = (unsigned int*)row;
      constexpr int U = NXP*CIN/2;
      for (int idx = lane; idx < U; idx += 32) {
        int p  = idx*2;
        int xp = p / CIN, c = p % CIN;
        int x  = x0 + xp;
        unsigned int val = 0u;
        if (x >= 0 && x < WIN) val = src[(x*CIN + c) >> 1];
        dstu[idx] = val;
      }
      __syncthreads();

      const v4u* lp = (const v4u*)row;
      const size_t wrow = (size_t)(kz*KH + ky)*COUT;
#pragma unroll
      for (int kb = 0; kb < KPAD; kb += 32) {
        // A fragments, one per m-tile (ISA 16-bit A layout: lane half picks K block
        // of 8; bf16 pairs per VGPR). A(m,k) == row[m*SX*CIN + k], k = kx*CIN + c.
        FragB16 a[MT];
#pragma unroll
        for (int mt = 0; mt < MT; ++mt) {
          const int abase = (mt*16 + lo)*SX*CIN;
          a[mt].q[0] = lp[(abase + kb + hi*8) >> 3];
          a[mt].q[1] = lp[(abase + kb + 16 + hi*8) >> 3];
        }
#pragma unroll
        for (int nt = 0; nt < NT; ++nt) {
          // B fragment (same K order as A: lane half picks K block of 8; col = lane&15)
          FragB16 bm;
          const v4u* wp = (const v4u*)(wb + (wrow + nt*16 + lo)*KPAD + kb + hi*8);
          bm.q[0] = wp[0];
          bm.q[1] = wp[2];
#pragma unroll
          for (int mt = 0; mt < MT; ++mt) {
            acc[mt][nt] = __builtin_amdgcn_wmma_f32_16x16x32_bf16(
                false, a[mt].v, false, bm.v, (short)0, acc[mt][nt], false, false);
          }
        }
      }
      __syncthreads();
    }
  }

  // epilogue: fused BN (scale, bias), ReLU, mask multiply, bf16 store
  const size_t obase = (size_t)((b*DOUT + zo)*HOUT + yo) * WOUT;
#pragma unroll
  for (int nt = 0; nt < NT; ++nt) {
    const int co = nt*16 + lo;
    const float s = scale[co], bb = bias[co];
#pragma unroll
    for (int mt = 0; mt < MT; ++mt) {
#pragma unroll
      for (int r = 0; r < 8; ++r) {
        const int x = (xtw*MT + mt)*16 + hi*8 + r;  // C/D layout: VGPR r -> M = hi*8 + r
        if (x < WOUT) {
          float v = acc[mt][nt][r]*s + bb;
          v = v > 0.0f ? v : 0.0f;
          v *= (float)omask[obase + x];
          act_out[(obase + x)*COUT + co] = (__bf16)v;
        }
      }
    }
  }
}

// ---------------- final transpose: (B,D,H,W,C) bf16 -> (B, C*D, H, W) f32 ----------------
__global__ void write_out_kernel(const __bf16* __restrict__ act4, float* __restrict__ out)
{
  const int total = 2*2*25*22*128;
  int t = blockIdx.x * blockDim.x + threadIdx.x;
  if (t >= total) return;
  int c = t % 128;
  int w = (t / 128) % 22;
  int h = (t / (128*22)) % 25;
  int d = (t / (128*22*25)) % 2;
  int b = t / (128*22*25*2);
  out[(size_t)(((b*128 + c)*2 + d)*25 + h)*22 + w] = (float)act4[t];
}

extern "C" void kernel_launch(void* const* d_in, const int* in_sizes, int n_in,
                              void* d_out, int out_size, void* d_ws, size_t ws_size,
                              hipStream_t stream)
{
  (void)n_in; (void)out_size; (void)ws_size;
  const float* pts   = (const float*)d_in[0];
  const float* feats = (const float*)d_in[1];
  const int npts = in_sizes[0] / 4;

  // params may be flattened insertion-order (w,s,b) or alphabetical (b,s,w); s is at 14 either way
  const int wbase = (in_sizes[2] > 1000) ? 2 : 26;
  const int bbase = (wbase == 2) ? 26 : 2;
  const float *W[12], *S[12], *Bp[12];
  for (int i = 0; i < 12; ++i) {
    W[i]  = (const float*)d_in[wbase + i];
    S[i]  = (const float*)d_in[14 + i];
    Bp[i] = (const float*)d_in[bbase + i];
  }

  char* ws = (char*)d_ws;
  size_t off = 0;
  auto alloc = [&off](size_t bytes) {
    size_t o = off; off = (off + bytes + 255) & ~(size_t)255; return o;
  };

  const size_t o_sums  = alloc((size_t)NV0*16*4);   // i32 sums; region reused for act0b bf16
  const size_t o_cnt   = alloc((size_t)NV0*4);
  const size_t o_act0a = alloc((size_t)NV0*16*2);
  const size_t o_mask0 = alloc((size_t)NV0);
  const size_t o_act1a = alloc((size_t)NV1*32*2);
  const size_t o_act1b = alloc((size_t)NV1*32*2);
  const size_t o_mask1 = alloc((size_t)NV1);
  const size_t o_act2a = alloc((size_t)NV2*64*2);
  const size_t o_act2b = alloc((size_t)NV2*64*2);
  const size_t o_mask2 = alloc((size_t)NV2);
  const size_t o_act3a = alloc((size_t)NV3*64*2);
  const size_t o_act3b = alloc((size_t)NV3*64*2);
  const size_t o_mask3 = alloc((size_t)NV3);
  const size_t o_act4  = alloc((size_t)NV4*128*2);
  const size_t o_mask4 = alloc((size_t)NV4);

  const int wsz[12] = {9216, 9216, 18432, 27648, 27648, 55296,
                       110592, 110592, 110592, 110592, 110592, 24576};
  size_t o_w[12];
  for (int i = 0; i < 12; ++i) o_w[i] = alloc((size_t)wsz[i]*2);

  int*           sums  = (int*)(ws + o_sums);
  unsigned int*  cnt   = (unsigned int*)(ws + o_cnt);
  __bf16* act0a = (__bf16*)(ws + o_act0a);
  __bf16* act0b = (__bf16*)(ws + o_sums);   // overlays dead sums region
  __bf16* act1a = (__bf16*)(ws + o_act1a);
  __bf16* act1b = (__bf16*)(ws + o_act1b);
  __bf16* act2a = (__bf16*)(ws + o_act2a);
  __bf16* act2b = (__bf16*)(ws + o_act2b);
  __bf16* act3a = (__bf16*)(ws + o_act3a);
  __bf16* act3b = (__bf16*)(ws + o_act3b);
  __bf16* act4  = (__bf16*)(ws + o_act4);
  unsigned char* mask0 = (unsigned char*)(ws + o_mask0);
  unsigned char* mask1 = (unsigned char*)(ws + o_mask1);
  unsigned char* mask2 = (unsigned char*)(ws + o_mask2);
  unsigned char* mask3 = (unsigned char*)(ws + o_mask3);
  unsigned char* mask4 = (unsigned char*)(ws + o_mask4);
  __bf16* wb[12];
  for (int i = 0; i < 12; ++i) wb[i] = (__bf16*)(ws + o_w[i]);

  hipMemsetAsync(sums, 0, (size_t)NV0*16*4, stream);
  hipMemsetAsync(cnt,  0, (size_t)NV0*4,  stream);

  // repack weights to padded bf16
  prep_w<16,16,3,3,3><<<(9216+255)/256, 256, 0, stream>>>(W[0], wb[0]);
  prep_w<16,16,3,3,3><<<(9216+255)/256, 256, 0, stream>>>(W[1], wb[1]);
  prep_w<16,32,3,3,3><<<(18432+255)/256, 256, 0, stream>>>(W[2], wb[2]);
  prep_w<32,32,3,3,3><<<(27648+255)/256, 256, 0, stream>>>(W[3], wb[3]);
  prep_w<32,32,3,3,3><<<(27648+255)/256, 256, 0, stream>>>(W[4], wb[4]);
  prep_w<32,64,3,3,3><<<(55296+255)/256, 256, 0, stream>>>(W[5], wb[5]);
  prep_w<64,64,3,3,3><<<(110592+255)/256, 256, 0, stream>>>(W[6], wb[6]);
  prep_w<64,64,3,3,3><<<(110592+255)/256, 256, 0, stream>>>(W[7], wb[7]);
  prep_w<64,64,3,3,3><<<(110592+255)/256, 256, 0, stream>>>(W[8], wb[8]);
  prep_w<64,64,3,3,3><<<(110592+255)/256, 256, 0, stream>>>(W[9], wb[9]);
  prep_w<64,64,3,3,3><<<(110592+255)/256, 256, 0, stream>>>(W[10], wb[10]);
  prep_w<64,128,3,1,1><<<(24576+255)/256, 256, 0, stream>>>(W[11], wb[11]);

  // voxelize + finalize
  voxelize_kernel<<<((size_t)npts*16 + 255)/256, 256, 0, stream>>>(pts, feats, sums, cnt, npts);
  finalize_kernel<<<((size_t)NV0*16 + 255)/256, 256, 0, stream>>>(sums, cnt, act0a, mask0, NV0);

  // L1, L2: subm 16->16 @ res0  (XT=11 -> MT=4, 3 wave-tiles; 72 WMMA/wave)
  {
    dim3 g(3, 41*200, 2);
    conv_wmma<16,16,3,3,3,1,1,1,1,1,1,41,200,176,41,200,176,4>
      <<<g, 32, 0, stream>>>(act0a, act0b, wb[0], S[0], Bp[0], mask0);
    conv_wmma<16,16,3,3,3,1,1,1,1,1,1,41,200,176,41,200,176,4>
      <<<g, 32, 0, stream>>>(act0b, act0a, wb[1], S[1], Bp[1], mask0);
  }
  // L3: sp 16->32 stride 2 -> res1  (XT=6 -> MT=2)
  mask_conv<3,3,3,2,2,2,1,1,1,41,200,176,21,100,88>
    <<<(NV1+255)/256, 256, 0, stream>>>(mask0, mask1);
  {
    dim3 g(3, 21*100, 2);
    conv_wmma<16,32,3,3,3,2,2,2,1,1,1,41,200,176,21,100,88,2>
      <<<g, 32, 0, stream>>>(act0a, act1a, wb[2], S[2], Bp[2], mask1);
  }
  // L4, L5: subm 32->32 @ res1  (MT=2, NT=2)
  {
    dim3 g(3, 21*100, 2);
    conv_wmma<32,32,3,3,3,1,1,1,1,1,1,21,100,88,21,100,88,2>
      <<<g, 32, 0, stream>>>(act1a, act1b, wb[3], S[3], Bp[3], mask1);
    conv_wmma<32,32,3,3,3,1,1,1,1,1,1,21,100,88,21,100,88,2>
      <<<g, 32, 0, stream>>>(act1b, act1a, wb[4], S[4], Bp[4], mask1);
  }
  // L6: sp 32->64 stride 2 -> res2  (XT=3 -> MT=2, 2 wave-tiles, NT=4)
  mask_conv<3,3,3,2,2,2,1,1,1,21,100,88,11,50,44>
    <<<(NV2+255)/256, 256, 0, stream>>>(mask1, mask2);
  {
    dim3 g(2, 11*50, 2);
    conv_wmma<32,64,3,3,3,2,2,2,1,1,1,21,100,88,11,50,44,2>
      <<<g, 32, 0, stream>>>(act1a, act2a, wb[5], S[5], Bp[5], mask2);
  }
  // L7, L8: subm 64->64 @ res2  (MT=2, NT=4)
  {
    dim3 g(2, 11*50, 2);
    conv_wmma<64,64,3,3,3,1,1,1,1,1,1,11,50,44,11,50,44,2>
      <<<g, 32, 0, stream>>>(act2a, act2b, wb[6], S[6], Bp[6], mask2);
    conv_wmma<64,64,3,3,3,1,1,1,1,1,1,11,50,44,11,50,44,2>
      <<<g, 32, 0, stream>>>(act2b, act2a, wb[7], S[7], Bp[7], mask2);
  }
  // L9: sp 64->64 stride 2, pad (0,1,1) -> res3  (XT=2 -> MT=2, 1 wave-tile)
  mask_conv<3,3,3,2,2,2,0,1,1,11,50,44,5,25,22>
    <<<(NV3+255)/256, 256, 0, stream>>>(mask2, mask3);
  {
    dim3 g(1, 5*25, 2);
    conv_wmma<64,64,3,3,3,2,2,2,0,1,1,11,50,44,5,25,22,2>
      <<<g, 32, 0, stream>>>(act2a, act3a, wb[8], S[8], Bp[8], mask3);
  }
  // L10, L11: subm 64->64 @ res3  (MT=2, 1 wave-tile)
  {
    dim3 g(1, 5*25, 2);
    conv_wmma<64,64,3,3,3,1,1,1,1,1,1,5,25,22,5,25,22,2>
      <<<g, 32, 0, stream>>>(act3a, act3b, wb[9], S[9], Bp[9], mask3);
    conv_wmma<64,64,3,3,3,1,1,1,1,1,1,5,25,22,5,25,22,2>
      <<<g, 32, 0, stream>>>(act3b, act3a, wb[10], S[10], Bp[10], mask3);
  }
  // L12: sp 64->128, k=(3,1,1), stride (2,1,1), pad 0 -> res4  (MT=2, NT=8)
  mask_conv<3,1,1,2,1,1,0,0,0,5,25,22,2,25,22>
    <<<(NV4+255)/256, 256, 0, stream>>>(mask3, mask4);
  {
    dim3 g(1, 2*25, 2);
    conv_wmma<64,128,3,1,1,2,1,1,0,0,0,5,25,22,2,25,22,2>
      <<<g, 32, 0, stream>>>(act3a, act4, wb[11], S[11], Bp[11], mask4);
  }

  // transpose to (B, C*D, H, W) f32
  write_out_kernel<<<(NV4*128 + 255)/256, 256, 0, stream>>>(act4, (float*)d_out);
}
